// SelfAttention_4458176053819
// MI455X (gfx1250) — matrix operand order repaired and checked by reading, not compiled, not run on hardware
//
#include <hip/hip_runtime.h>
#include <hip/hip_bf16.h>
#include <stddef.h>
#include <stdint.h>

// ---------------------------------------------------------------------------
// Problem constants (reference: B=2, S=2048, D=1024, H=16, hd=64)
// ---------------------------------------------------------------------------
#define BATCH 2
#define SEQ   2048
#define DIM   1024
#define HEADS 16
#define HDIM  64
#define MROWS (BATCH * SEQ)          // 4096
#define LN_EPSF 1e-5f

// ---------------------------------------------------------------------------
// Vector types (own typedefs: HIP's uint4/float4 have ctors -> bad in unions)
// ---------------------------------------------------------------------------
typedef __attribute__((ext_vector_type(16))) __bf16 bf16x16;
typedef __attribute__((ext_vector_type(8)))  float  f32x8;
typedef __attribute__((ext_vector_type(4)))  float  f32x4;
typedef __attribute__((ext_vector_type(4)))  unsigned int u32x4;
typedef __attribute__((ext_vector_type(8)))  int    i32x8;
typedef __attribute__((ext_vector_type(4)))  int    i32x4;

union FragB {            // one 16x32 A-frag or 32x16 B-frag of bf16 (32 bytes)
    u32x4          q[2];
    bf16x16        v;
    unsigned short us[16];
};

static __device__ __forceinline__ unsigned short f2bf(float f) {
    unsigned u = __builtin_bit_cast(unsigned, f);
    return (unsigned short)((u + 0x7FFFu + ((u >> 16) & 1u)) >> 16); // RNE
}

static __device__ __forceinline__ f32x8 wmma_bf16(bf16x16 a, bf16x16 b, f32x8 c) {
    // v_wmma_f32_16x16x32_bf16 (neg_a, A, neg_b, B, c_mod, C, reuse_a, reuse_b)
    return __builtin_amdgcn_wmma_f32_16x16x32_bf16(false, a, false, b, (short)0, c,
                                                   false, false);
}

// ---------------------------------------------------------------------------
// 16-lane (DPP row) xor reductions: VALU-only, no LDS round-trips.
// ROW_XMASK ctrl = 0x160 | xmask (xor within a row of 16 lanes).
// ---------------------------------------------------------------------------
#if defined(__has_builtin)
#if __has_builtin(__builtin_amdgcn_update_dpp)
#define HAVE_DPP 1
#endif
#endif

#ifdef HAVE_DPP
template <int CTRL>
static __device__ __forceinline__ float dpp_xor(float x) {
    return __builtin_bit_cast(float,
        __builtin_amdgcn_update_dpp(0, __builtin_bit_cast(int, x),
                                    CTRL, 0xf, 0xf, true));
}
static __device__ __forceinline__ float row16_max(float x) {
    x = fmaxf(x, dpp_xor<0x161>(x));   // xor 1
    x = fmaxf(x, dpp_xor<0x162>(x));   // xor 2
    x = fmaxf(x, dpp_xor<0x164>(x));   // xor 4
    x = fmaxf(x, dpp_xor<0x168>(x));   // xor 8
    return x;
}
static __device__ __forceinline__ float row16_sum(float x) {
    x += dpp_xor<0x161>(x);
    x += dpp_xor<0x162>(x);
    x += dpp_xor<0x164>(x);
    x += dpp_xor<0x168>(x);
    return x;
}
#else
static __device__ __forceinline__ float row16_max(float x) {
    for (int o = 1; o < 16; o <<= 1) x = fmaxf(x, __shfl_xor(x, o, 16));
    return x;
}
static __device__ __forceinline__ float row16_sum(float x) {
    for (int o = 1; o < 16; o <<= 1) x += __shfl_xor(x, o, 16);
    return x;
}
#endif

// ---------------------------------------------------------------------------
// Tensor Data Mover: 2D tile Global -> LDS (CDNA5 TDM, TENSORcnt-tracked).
// D# per CDNA5 ISA. This toolchain's builtin is the
// 6-arg form: (g0 u32x4, g1 i32x8, g2 i32x4, g3 i32x4, g4 i32x8, cpol i32).
// ---------------------------------------------------------------------------
#if defined(__has_builtin)
#if __has_builtin(__builtin_amdgcn_tensor_load_to_lds) && \
    __has_builtin(__builtin_amdgcn_s_wait_tensorcnt)
#define USE_TDM 1
#endif
#endif

#ifdef USE_TDM
static __device__ __forceinline__ unsigned lds_addr_of(const void* p) {
    // LDS aperture: flat addr low 32 bits are the LDS byte address (ISA 10.2)
    return (unsigned)(uintptr_t)p;
}

// Load tile_d1 rows x tile_d0 elems (2-byte elems) from row-major tensor with
// row stride `stride0` (elems) at `gsrc` into LDS at byte offset `lds_addr`.
static __device__ __forceinline__ void tdm_load_2d_bf16(unsigned lds_addr,
                                                        const void* gsrc,
                                                        unsigned tile_d0,
                                                        unsigned tile_d1,
                                                        unsigned long long stride0) {
    const unsigned long long ga = (unsigned long long)(uintptr_t)gsrc;
    u32x4 g0;
    g0[0] = 1u;                                       // count=1 (valid), user
    g0[1] = lds_addr;                                 // lds_addr [63:32]
    g0[2] = (unsigned)(ga & 0xffffffffu);             // global_addr[31:0]
    g0[3] = (unsigned)((ga >> 32) & 0x01ffffffu)      // global_addr[56:32]
          | (2u << 30);                               // type=2 ("image")
    const unsigned tensor_d0 = tile_d0;               // exact-fit tensor
    const unsigned tensor_d1 = tile_d1;
    i32x8 g1;
    g1[0] = (int)(1u << 16);                          // data_size=1 (2 bytes)
    g1[1] = (int)((tensor_d0 & 0xffffu) << 16);       // tensor_dim0[15:0]
    g1[2] = (int)(((tensor_d0 >> 16) & 0xffffu) |
                  ((tensor_d1 & 0xffffu) << 16));     // dim0[31:16], dim1[15:0]
    g1[3] = (int)(((tensor_d1 >> 16) & 0xffffu) |
                  ((tile_d0 & 0xffffu) << 16));       // dim1[31:16], tile_dim0
    g1[4] = (int)(tile_d1 & 0xffffu);                 // tile_dim1 (tile_dim2=0)
    g1[5] = (int)(unsigned)(stride0 & 0xffffffffu);   // dim0_stride[31:0]
    g1[6] = (int)(unsigned)((stride0 >> 32) & 0xffffu);
    g1[7] = 0;
    i32x4 g2; g2[0] = 0; g2[1] = 0; g2[2] = 0; g2[3] = 0;
    i32x4 g3 = g2;
    i32x8 g4;
#pragma unroll
    for (int i = 0; i < 8; ++i) g4[i] = 0;
    __builtin_amdgcn_tensor_load_to_lds(g0, g1, g2, g3, g4, 0);
}
#endif

// ---------------------------------------------------------------------------
// Kernel 0: fp32 -> bf16 weight conversion (weights then live in 192MB L2)
// ---------------------------------------------------------------------------
__global__ void cvt_f32_bf16_kernel(const float* __restrict__ in,
                                    unsigned short* __restrict__ out, int n) {
    int i = blockIdx.x * blockDim.x + threadIdx.x;
    if (i < n) out[i] = f2bf(in[i]);
}

// ---------------------------------------------------------------------------
// Kernel 1/3: WMMA GEMM   Y[M,N] = (X[M,K]f32 @ W[N,K]^T_bf16 + bias)
//   BF16OUT=true : Y = bf16( (acc)*scale )        bias folded into C-init
//   BF16OUT=false: Y = f32( acc + residual )
// Each wave: one 16-row M tile x four 16-col N tiles. blockDim=128 (4 waves).
// grid = (N/64, M/64).
// ---------------------------------------------------------------------------
template <bool BF16OUT>
__global__ void wmma_gemm_kernel(const float* __restrict__ X,
                                 const unsigned short* __restrict__ W,
                                 const float* __restrict__ bias,
                                 const float* __restrict__ residual,
                                 float scale,
                                 unsigned short* __restrict__ Yb,
                                 float* __restrict__ Yf,
                                 int M, int N, int K) {
    const int lane = threadIdx.x & 31;
    const int wv   = threadIdx.x >> 5;      // 0..3
    const int half = lane >> 4;             // K-half select (ISA A/B layouts)
    const int l16  = lane & 15;

    const int mTile = blockIdx.y * 4 + wv;
    const int nBase = blockIdx.x * 64;
    const int row0  = mTile * 16;

    // C-init: bias broadcast down each output column
    f32x8 acc[4];
#pragma unroll
    for (int n = 0; n < 4; ++n) {
        const float bb = bias[nBase + n * 16 + l16];
#pragma unroll
        for (int v = 0; v < 8; ++v) acc[n][v] = bb;
    }

    const float* xrow = X + (size_t)(row0 + l16) * K;

    for (int k0 = 0; k0 < K; k0 += 32) {
        // A frag: lane row=l16; e<8 -> k=k0+half*8+e; e>=8 -> k0+16+half*8+(e-8)
        FragB a;
        const f32x4* xp0 = reinterpret_cast<const f32x4*>(xrow + k0 + half * 8);
        const f32x4* xp1 = reinterpret_cast<const f32x4*>(xrow + k0 + 16 + half * 8);
        f32x4 c0 = xp0[0], c1 = xp0[1], c2 = xp1[0], c3 = xp1[1];
        float tf[16] = {c0.x, c0.y, c0.z, c0.w, c1.x, c1.y, c1.z, c1.w,
                        c2.x, c2.y, c2.z, c2.w, c3.x, c3.y, c3.z, c3.w};
#pragma unroll
        for (int e = 0; e < 16; ++e) a.us[e] = f2bf(tf[e]);

        // B frags: col n = l16; e<8 -> k=k0+half*8+e; e>=8 -> k0+16+half*8+(e-8)
#pragma unroll
        for (int n = 0; n < 4; ++n) {
            const u32x4* wp = reinterpret_cast<const u32x4*>(
                W + (size_t)(nBase + n * 16 + l16) * K + k0);
            FragB b;
            b.q[0] = wp[half];
            b.q[1] = wp[2 + half];
            acc[n] = wmma_bf16(a.v, b.v, acc[n]);
        }
    }

    // epilogue: C layout: lane holds (row = v + 8*half, col = l16)
#pragma unroll
    for (int n = 0; n < 4; ++n) {
        const int col = nBase + n * 16 + l16;
#pragma unroll
        for (int v = 0; v < 8; ++v) {
            const int r = row0 + v + 8 * half;
            if (BF16OUT) {
                Yb[(size_t)r * N + col] = f2bf(acc[n][v] * scale);
            } else {
                Yf[(size_t)r * N + col] = acc[n][v] + residual[(size_t)r * N + col];
            }
        }
    }
}

// ---------------------------------------------------------------------------
// Kernel 2: flash attention.
//   Q/K/V: bf16 [B*S, D] (head h occupies cols h*64..h*64+63)
//   blockDim=256 (8 waves); all 8 waves share (b,h). K/V 32x64 tiles are
//   double-buffered in LDS and streamed by the TDM (tensor_load_to_lds) one
//   tile ahead of compute; each wave owns a 16-query tile.
//   grid = (S/128, B*H).
// ---------------------------------------------------------------------------
__global__ void flash_attn_kernel(const unsigned short* __restrict__ Q,
                                  const unsigned short* __restrict__ Km,
                                  const unsigned short* __restrict__ V,
                                  const float* __restrict__ extra,  // [B*H,S,S]
                                  const float* __restrict__ mask,   // [B,1,S,S]
                                  float* __restrict__ O) {          // [B*S,D] f32
    __shared__ unsigned short kstage[2][32][64];   // K tile  [buf][key][d]
    __shared__ unsigned short vstage[2][32][64];   // V tile  [buf][key][d]
    __shared__ unsigned short pstage[8][16][40];   // per-wave P 16x32 (pad->40)

    const int tid  = threadIdx.x;
    const int lane = tid & 31;
    const int wv   = tid >> 5;   // 0..7
    const int half = lane >> 4;
    const int l16  = lane & 15;

    const int bh = blockIdx.y;
    const int b  = bh >> 4;              // H == 16
    const int h  = bh & 15;
    const int s0 = (blockIdx.x * 8 + wv) * 16;   // query tile base (seq pos)

    // ---- Q A-fragments (16 queries x 64 dims -> two 16x32 frags)
    const u32x4* qp = reinterpret_cast<const u32x4*>(
        Q + (size_t)(b * SEQ + s0 + l16) * DIM + h * HDIM);
    FragB qa0, qa1;
    qa0.q[0] = qp[half];     qa0.q[1] = qp[2 + half];
    qa1.q[0] = qp[4 + half]; qa1.q[1] = qp[6 + half];

    float m[8], ell[8];
#pragma unroll
    for (int v = 0; v < 8; ++v) { m[v] = -1e30f; ell[v] = 0.f; }
    f32x8 acc[4] = {};

    const float* exrow = extra + (size_t)bh * SEQ * SEQ;
    const float* mkrow = mask + (size_t)b * SEQ * SEQ;
    const unsigned short* Kg = Km + (size_t)(b * SEQ) * DIM + h * HDIM;
    const unsigned short* Vg = V + (size_t)(b * SEQ) * DIM + h * HDIM;

#ifdef USE_TDM
    // ---- prologue: TDM-stream tile 0 into buffer 0
    if (tid < 32) {
        tdm_load_2d_bf16(lds_addr_of(&kstage[0][0][0]), Kg, 64u, 32u,
                         (unsigned long long)DIM);
        tdm_load_2d_bf16(lds_addr_of(&vstage[0][0][0]), Vg, 64u, 32u,
                         (unsigned long long)DIM);
    }
#endif

    for (int key0 = 0; key0 < SEQ; key0 += 32) {
#ifdef USE_TDM
        const int cur = (key0 >> 5) & 1;
        if (tid < 32) __builtin_amdgcn_s_wait_tensorcnt(0);  // tile `cur` done
        // barrier: publishes tile `cur` AND proves last readers of buf cur^1
        // (iteration key0-32) are done before the TDM below overwrites it
        __syncthreads();
        if (tid < 32 && key0 + 32 < SEQ) {
            tdm_load_2d_bf16(lds_addr_of(&kstage[cur ^ 1][0][0]),
                             Kg + (size_t)(key0 + 32) * DIM, 64u, 32u,
                             (unsigned long long)DIM);
            tdm_load_2d_bf16(lds_addr_of(&vstage[cur ^ 1][0][0]),
                             Vg + (size_t)(key0 + 32) * DIM, 64u, 32u,
                             (unsigned long long)DIM);
        }
#else
        const int cur = 0;
        __syncthreads();   // prior readers done before overwrite
        {
            const int row = tid >> 3;           // 0..31
            const int c8  = (tid & 7) * 8;      // 8 bf16 = 16B per thread
            *reinterpret_cast<u32x4*>(&kstage[0][row][c8]) =
                *reinterpret_cast<const u32x4*>(Kg + (size_t)(key0 + row) * DIM + c8);
            *reinterpret_cast<u32x4*>(&vstage[0][row][c8]) =
                *reinterpret_cast<const u32x4*>(Vg + (size_t)(key0 + row) * DIM + c8);
        }
        __syncthreads();
#endif
        // prefetch upcoming extra_attn rows (dominant HBM stream)
        __builtin_prefetch(exrow + (size_t)(s0 + half * 8) * SEQ + key0 + 32, 0, 0);

        // ---- batch-load extra+mask for both 16-key sub-tiles (read-once: NT)
        f32x8 ex0, ex1, mk0, mk1;
#pragma unroll
        for (int v = 0; v < 8; ++v) {
            const size_t qoff = (size_t)(s0 + v + 8 * half) * SEQ + key0 + l16;
            ex0[v] = __builtin_nontemporal_load(exrow + qoff);
            ex1[v] = __builtin_nontemporal_load(exrow + qoff + 16);
        }
#pragma unroll
        for (int v = 0; v < 8; ++v) {
            const size_t qoff = (size_t)(s0 + v + 8 * half) * SEQ + key0 + l16;
            mk0[v] = __builtin_nontemporal_load(mkrow + qoff);
            mk1[v] = __builtin_nontemporal_load(mkrow + qoff + 16);
        }
        const f32x8 cin0 = ex0 + mk0;
        const f32x8 cin1 = ex1 + mk1;

        // ---- scores: two 16x16 tiles via WMMA, C preloaded with extra+mask
        f32x8 sc[2];
#pragma unroll
        for (int j = 0; j < 2; ++j) {
            const u32x4* kp0 = reinterpret_cast<const u32x4*>(
                &kstage[cur][j * 16 + l16][0]);
            const u32x4* kp1 = reinterpret_cast<const u32x4*>(
                &kstage[cur][j * 16 + l16][32]);
            FragB bk0, bk1;
            bk0.q[0] = kp0[half]; bk0.q[1] = kp0[2 + half];
            bk1.q[0] = kp1[half]; bk1.q[1] = kp1[2 + half];
            sc[j] = wmma_bf16(qa1.v, bk1.v,
                              wmma_bf16(qa0.v, bk0.v, j ? cin1 : cin0));
        }

        // ---- online softmax across the 32-key pair; row reductions use
        //      DPP16 ROW_XMASK (pure VALU, no LDS round-trips)
#pragma unroll
        for (int v = 0; v < 8; ++v) {
            const float mx = row16_max(fmaxf(sc[0][v], sc[1][v]));
            const float mn = fmaxf(m[v], mx);
            const float al = __expf(m[v] - mn);
            m[v] = mn;
            const float p0 = __expf(sc[0][v] - mn);
            const float p1 = __expf(sc[1][v] - mn);
            const float rs = row16_sum(p0 + p1);
            ell[v] = ell[v] * al + rs;
            acc[0][v] *= al; acc[1][v] *= al; acc[2][v] *= al; acc[3][v] *= al;
            // stage P (C-layout) into per-wave LDS slab as [row][key] bf16
            pstage[wv][v + 8 * half][l16]      = f2bf(p0);
            pstage[wv][v + 8 * half][16 + l16] = f2bf(p1);
        }

        // ---- read P back in A-layout (row = l16; keys half*16..half*16+15)
        const u32x4* lp = reinterpret_cast<const u32x4*>(&pstage[wv][l16][0]);
        FragB pa;
        pa.q[0] = lp[2 * half];
        pa.q[1] = lp[2 * half + 1];

        // ---- P @ V : four 16-col d tiles, V B-frags from LDS
#pragma unroll
        for (int n = 0; n < 4; ++n) {
            FragB bv;
#pragma unroll
            for (int e = 0; e < 16; ++e) {
                bv.us[e] = vstage[cur][half * 16 + e][n * 16 + l16];
            }
            acc[n] = wmma_bf16(pa.v, bv.v, acc[n]);
        }
        // (TDM path: next iteration's barrier separates these reads from the
        //  next overwrite of buf `cur`; fallback path barriers explicitly)
    }

    // ---- epilogue: O[b*S + q, h*64 + d] = acc / ell
#pragma unroll
    for (int n = 0; n < 4; ++n) {
#pragma unroll
        for (int v = 0; v < 8; ++v) {
            const int qs = s0 + v + 8 * half;
            O[(size_t)(b * SEQ + qs) * DIM + h * HDIM + n * 16 + l16] =
                acc[n][v] / ell[v];
        }
    }
}

// ---------------------------------------------------------------------------
// Kernel 4: row LayerNorm over D=1024. One block (256 thr) per row.
// ---------------------------------------------------------------------------
__global__ void layernorm_kernel(const float* __restrict__ y,
                                 const float* __restrict__ gamma,
                                 const float* __restrict__ beta,
                                 float* __restrict__ out) {
    const int row = blockIdx.x;
    const int tid = threadIdx.x;
    const float* yr = y + (size_t)row * DIM;

    // cache this thread's 4 elements in registers
    float vals[4];
    float s = 0.f, s2 = 0.f;
#pragma unroll
    for (int c = 0; c < 4; ++c) {
        vals[c] = yr[tid + 256 * c];
        s += vals[c]; s2 += vals[c] * vals[c];
    }
#pragma unroll
    for (int o = 1; o < 32; o <<= 1) {
        s  += __shfl_xor(s, o, 32);
        s2 += __shfl_xor(s2, o, 32);
    }
    __shared__ float shs[8], sh2[8];
    if ((tid & 31) == 0) { shs[tid >> 5] = s; sh2[tid >> 5] = s2; }
    __syncthreads();
    s = 0.f; s2 = 0.f;
#pragma unroll
    for (int w = 0; w < 8; ++w) { s += shs[w]; s2 += sh2[w]; }

    const float mu  = s * (1.0f / DIM);
    const float var = s2 * (1.0f / DIM) - mu * mu;
    const float inv = rsqrtf(var + LN_EPSF);
#pragma unroll
    for (int c = 0; c < 4; ++c) {
        const int i = tid + 256 * c;
        __builtin_nontemporal_store((vals[c] - mu) * inv * gamma[i] + beta[i],
                                    out + (size_t)row * DIM + i);
    }
}

// ---------------------------------------------------------------------------
// Host launcher
// ---------------------------------------------------------------------------
extern "C" void kernel_launch(void* const* d_in, const int* in_sizes, int n_in,
                              void* d_out, int out_size, void* d_ws, size_t ws_size,
                              hipStream_t stream) {
    (void)in_sizes; (void)n_in; (void)out_size; (void)ws_size;

    const float* hidden = (const float*)d_in[0];
    const float* amask  = (const float*)d_in[1];
    const float* extra  = (const float*)d_in[2];
    const float* Wq = (const float*)d_in[3];  const float* bq = (const float*)d_in[4];
    const float* Wk = (const float*)d_in[5];  const float* bk = (const float*)d_in[6];
    const float* Wv = (const float*)d_in[7];  const float* bv = (const float*)d_in[8];
    const float* Wo = (const float*)d_in[9];  const float* bo = (const float*)d_in[10];
    const float* lng = (const float*)d_in[11];
    const float* lnb = (const float*)d_in[12];
    float* out = (float*)d_out;

    // workspace layout (bytes): 4x2MB bf16 weights, 3x8MB bf16 QKV,
    // 16MB f32 attn, 16MB f32 pre-LN  => 64MB total
    char* ws = (char*)d_ws;
    const size_t MB = 1024 * 1024;
    unsigned short* Wq_b = (unsigned short*)(ws + 0 * MB);
    unsigned short* Wk_b = (unsigned short*)(ws + 2 * MB);
    unsigned short* Wv_b = (unsigned short*)(ws + 4 * MB);
    unsigned short* Wo_b = (unsigned short*)(ws + 6 * MB);
    unsigned short* Qb   = (unsigned short*)(ws + 8 * MB);
    unsigned short* Kb   = (unsigned short*)(ws + 16 * MB);
    unsigned short* Vb   = (unsigned short*)(ws + 24 * MB);
    float*          attnF = (float*)(ws + 32 * MB);
    float*          yF    = (float*)(ws + 48 * MB);

    // 0) weight conversion
    {
        const int n = DIM * DIM;
        dim3 g((n + 255) / 256), bdim(256);
        cvt_f32_bf16_kernel<<<g, bdim, 0, stream>>>(Wq, Wq_b, n);
        cvt_f32_bf16_kernel<<<g, bdim, 0, stream>>>(Wk, Wk_b, n);
        cvt_f32_bf16_kernel<<<g, bdim, 0, stream>>>(Wv, Wv_b, n);
        cvt_f32_bf16_kernel<<<g, bdim, 0, stream>>>(Wo, Wo_b, n);
    }

    // 1) Q/K/V projections (WMMA bf16, f32 accumulate)
    {
        dim3 g(DIM / 64, MROWS / 64), bdim(128);
        const float qscale = 0.125f;  // hd^-0.5, hd = 64
        wmma_gemm_kernel<true><<<g, bdim, 0, stream>>>(hidden, Wq_b, bq, nullptr,
                                                       qscale, Qb, nullptr,
                                                       MROWS, DIM, DIM);
        wmma_gemm_kernel<true><<<g, bdim, 0, stream>>>(hidden, Wk_b, bk, nullptr,
                                                       1.0f, Kb, nullptr,
                                                       MROWS, DIM, DIM);
        wmma_gemm_kernel<true><<<g, bdim, 0, stream>>>(hidden, Wv_b, bv, nullptr,
                                                       1.0f, Vb, nullptr,
                                                       MROWS, DIM, DIM);
    }

    // 2) flash attention
    {
        dim3 g(SEQ / 128, BATCH * HEADS), bdim(256);
        flash_attn_kernel<<<g, bdim, 0, stream>>>(Qb, Kb, Vb, extra, amask, attnF);
    }

    // 3) output projection + residual
    {
        dim3 g(DIM / 64, MROWS / 64), bdim(128);
        wmma_gemm_kernel<false><<<g, bdim, 0, stream>>>(attnF, Wo_b, bo, hidden,
                                                        1.0f, nullptr, yF,
                                                        MROWS, DIM, DIM);
    }

    // 4) LayerNorm -> d_out
    {
        dim3 g(MROWS), bdim(256);
        layernorm_kernel<<<g, bdim, 0, stream>>>(yF, lng, lnb, out);
    }
}
